// QuantumWarpAttention_18829136625875
// MI455X (gfx1250) — matrix operand order repaired and checked by reading, not compiled, not run on hardware
//
#include <hip/hip_runtime.h>

typedef __attribute__((ext_vector_type(8)))  __bf16 v8bf;
typedef __attribute__((ext_vector_type(16))) __bf16 v16bf;
typedef __attribute__((ext_vector_type(8)))  float  v8f;
typedef __attribute__((ext_vector_type(4)))  int    v4i;

static __device__ __forceinline__ unsigned short f2bf(float f) {
    unsigned u = __builtin_bit_cast(unsigned, f);
    unsigned r = u + 0x7FFFu + ((u >> 16) & 1u);   // round-to-nearest-even
    return (unsigned short)(r >> 16);
}

// ---- CDNA5 async Global->LDS copy (16B per lane), tracked by ASYNCcnt ----
static __device__ __forceinline__ void async_copy16(const unsigned short* g,
                                                    unsigned short* l) {
#if __has_builtin(__builtin_amdgcn_global_load_async_to_lds_b128)
    __builtin_amdgcn_global_load_async_to_lds_b128(
        (__attribute__((address_space(1))) v4i*)g,
        (__attribute__((address_space(3))) v4i*)l, 0, 0);
#else
    // LDS address is the wave's LDS base plus the VGPR offset; low 32 bits of the flat LDS address are
    // the LDS byte offset (ISA 10.2).
    unsigned loff = (unsigned)(size_t)l;
    unsigned long long ga = (unsigned long long)(size_t)g;
    asm volatile("global_load_async_to_lds_b128 %0, %1, off"
                 :: "v"(loff), "v"(ga) : "memory");
#endif
}

static __device__ __forceinline__ void wait_async6() {
#if __has_builtin(__builtin_amdgcn_s_wait_asynccnt)
    __builtin_amdgcn_s_wait_asynccnt(6);
#else
    asm volatile("s_wait_asynccnt 0x6" ::: "memory");
#endif
}
static __device__ __forceinline__ void wait_async0() {
#if __has_builtin(__builtin_amdgcn_s_wait_asynccnt)
    __builtin_amdgcn_s_wait_asynccnt(0);
#else
    asm volatile("s_wait_asynccnt 0x0" ::: "memory");
#endif
}

// ---------------------------------------------------------------- converts
__global__ void convert_f32_bf16(const float* __restrict__ in,
                                 unsigned short* __restrict__ out, int n) {
    int i = blockIdx.x * blockDim.x + threadIdx.x;
    if (i < n) out[i] = f2bf(in[i]);
}

// W [K,N] f32 -> WT [N,K] bf16 (tiled transpose through LDS)
__global__ __launch_bounds__(256) void transpose_convert(const float* __restrict__ W,
                                                         unsigned short* __restrict__ WT,
                                                         int K, int N) {
    __shared__ float t[32][33];
    const int n0 = blockIdx.x * 32, k0 = blockIdx.y * 32;
    const int c = threadIdx.x & 31, rr = threadIdx.x >> 5;
#pragma unroll
    for (int p = 0; p < 4; ++p) {
        int r = rr + p * 8;
        t[r][c] = W[(size_t)(k0 + r) * N + n0 + c];
    }
    __syncthreads();
#pragma unroll
    for (int p = 0; p < 4; ++p) {
        int r = rr + p * 8;
        WT[(size_t)(n0 + r) * K + k0 + c] = f2bf(t[c][r]);
    }
}

// ---------------------------------------------------------------- bf16 WMMA GEMM
// C[M,N](f32) = A[M,K](bf16) * BT[N,K](bf16)^T + bias[N]
// block tile 128x64, BK=64, double-buffered LDS fed by async Global->LDS,
// 8 waves in 4x2, each wave 32x32 (2x2 WMMA 16x16x32), 8 WMMA per K-step.
#define AROW 72   // 64 K-halves + 8 pad
__global__ __launch_bounds__(256) void gemm_bf16_wmma(const unsigned short* __restrict__ A,
                                                      const unsigned short* __restrict__ BT,
                                                      const float* __restrict__ bias,
                                                      float* __restrict__ C,
                                                      int M, int N, int K) {
    __shared__ alignas(16) unsigned short As[2 * 128 * AROW];
    __shared__ alignas(16) unsigned short Bs[2 * 64 * AROW];

    const int tid  = threadIdx.x;
    const int lane = tid & 31;
    const int wave = tid >> 5;
    const int rowBase = (wave & 3) * 32;   // 4 waves along M
    const int colBase = (wave >> 2) * 32;  // 2 waves along N
    const int m0 = blockIdx.y * 128, n0 = blockIdx.x * 64;

    const int lm    = lane & 15;
    const int aksel = (lane < 16) ? 0 : 8;   // A frag: K 0..7/16..23 vs 8..15/24..31
    const int bksel = (lane < 16) ? 0 : 8;   // B frag: K 0..7/16..23 vs 8..15/24..31

    // stage one BK=64 slice: A 128x64 (4 chunks/thread), B 64x64 (2 chunks/thread)
    auto stage = [&](int buf, int k0) {
        unsigned short* ab = &As[buf * 128 * AROW];
        unsigned short* bb = &Bs[buf * 64 * AROW];
#pragma unroll
        for (int i = 0; i < 4; ++i) {
            int ch = tid + i * 256;
            int r = ch >> 3, q4 = ch & 7;
            async_copy16(&A[(size_t)(m0 + r) * K + k0 + q4 * 8], &ab[r * AROW + q4 * 8]);
        }
#pragma unroll
        for (int i = 0; i < 2; ++i) {
            int ch = tid + i * 256;
            int r = ch >> 3, q4 = ch & 7;
            async_copy16(&BT[(size_t)(n0 + r) * K + k0 + q4 * 8], &bb[r * AROW + q4 * 8]);
        }
    };

    v8f acc[2][2] = {};
    const int nK = K >> 6;

    stage(0, 0);
    if (nK > 1) stage(1, 64);

    for (int kt = 0; kt < nK; ++kt) {
        // the 6 youngest async ops belong to slice kt+1; async loads complete
        // in order, so asynccnt<=6 guarantees slice kt is fully in LDS.
        if (kt + 1 < nK) wait_async6(); else wait_async0();
        __syncthreads();

        const unsigned short* ab = &As[(kt & 1) * 128 * AROW];
        const unsigned short* bb = &Bs[(kt & 1) * 64 * AROW];

        if (kt + 3 < nK) {
            __builtin_prefetch(&A[(size_t)(m0 + (tid >> 1)) * K + (kt + 3) * 64], 0, 3);
            __builtin_prefetch(&BT[(size_t)(n0 + (tid & 63)) * K + (kt + 3) * 64], 0, 3);
        }

#pragma unroll
        for (int kk = 0; kk < 64; kk += 32) {
            v16bf af[2], bfr[2];
#pragma unroll
            for (int mi = 0; mi < 2; ++mi) {
                const unsigned short* ap = &ab[(rowBase + mi * 16 + lm) * AROW + kk + aksel];
                v8bf lo = *reinterpret_cast<const v8bf*>(ap);
                v8bf hi = *reinterpret_cast<const v8bf*>(ap + 16);
                af[mi] = __builtin_shufflevector(lo, hi, 0,1,2,3,4,5,6,7,8,9,10,11,12,13,14,15);
            }
#pragma unroll
            for (int ni = 0; ni < 2; ++ni) {
                const unsigned short* bp = &bb[(colBase + ni * 16 + lm) * AROW + kk + bksel];
                v8bf lo = *reinterpret_cast<const v8bf*>(bp);
                v8bf hi = *reinterpret_cast<const v8bf*>(bp + 16);
                bfr[ni] = __builtin_shufflevector(lo, hi, 0,1,2,3,4,5,6,7,8,9,10,11,12,13,14,15);
            }
#pragma unroll
            for (int mi = 0; mi < 2; ++mi)
#pragma unroll
                for (int ni = 0; ni < 2; ++ni)
                    acc[mi][ni] = __builtin_amdgcn_wmma_f32_16x16x32_bf16(
                        false, af[mi], false, bfr[ni], (short)0, acc[mi][ni], false, false);
        }
        __syncthreads();  // everyone done reading buf[kt&1]
        if (kt + 2 < nK) stage(kt & 1, (kt + 2) * 64);
    }

    // epilogue: lanes 0-15: N=lane, M=j; lanes 16-31: N=lane-16, M=8+j
#pragma unroll
    for (int mi = 0; mi < 2; ++mi)
#pragma unroll
        for (int ni = 0; ni < 2; ++ni) {
            int colg  = n0 + colBase + ni * 16 + lm;
            int rbase = m0 + rowBase + mi * 16 + ((lane >> 4) << 3);
            float bb2 = bias[colg];
#pragma unroll
            for (int j = 0; j < 8; ++j)
                C[(size_t)(rbase + j) * N + colg] = acc[mi][ni][j] + bb2;
        }
}

// ---------------------------------------------------------------- per-position attention
__global__ __launch_bounds__(256) void attn_kernel(const float* __restrict__ q,
                                                   const float* __restrict__ k,
                                                   const float* __restrict__ v,
                                                   const float* __restrict__ phase,
                                                   unsigned short* __restrict__ aout,
                                                   int S, int D) {
    const int m = blockIdx.x;
    const int b = m / S, s = m % S;
    const int tid = threadIdx.x;
    const size_t base = (size_t)m * 4 * D;

    float part[16];
#pragma unroll
    for (int j = 0; j < 16; ++j) part[j] = 0.f;

    for (int d = tid; d < D; d += 256) {
        float qr[4], qi[4], kr[4], ki[4];
#pragma unroll
        for (int h = 0; h < 4; ++h) {
            float ph = phase[h * D + d];
            float ch = __cosf(ph), sh = __sinf(ph);
            float qq = q[base + h * D + d];
            float kk = k[base + h * D + d];
            qr[h] = qq * ch; qi[h] = qq * sh;
            kr[h] = kk * ch; ki[h] = kk * sh;
        }
#pragma unroll
        for (int h = 0; h < 4; ++h)
#pragma unroll
            for (int g = 0; g < 4; ++g)
                part[h * 4 + g] += qr[h] * kr[g] + qi[h] * ki[g];
    }

    __shared__ float red[16 * 256];
#pragma unroll
    for (int j = 0; j < 16; ++j) red[j * 256 + tid] = part[j];
    __syncthreads();
    for (int off = 128; off > 0; off >>= 1) {
        if (tid < off) {
#pragma unroll
            for (int j = 0; j < 16; ++j)
                red[j * 256 + tid] += red[j * 256 + tid + off];
        }
        __syncthreads();
    }

    __shared__ float attnw[16];
    if (tid < 4) {
        const float scale = rsqrtf((float)D);
        const int h = tid;
        float sc[4], mx = -1e30f;
#pragma unroll
        for (int g = 0; g < 4; ++g) {
            sc[g] = red[(h * 4 + g) * 256] * scale;
            mx = fmaxf(mx, sc[g]);
        }
        float den = 0.f;
#pragma unroll
        for (int g = 0; g < 4; ++g) { sc[g] = __expf(sc[g] - mx); den += sc[g]; }
        float inv = 1.f / den;
#pragma unroll
        for (int g = 0; g < 4; ++g) attnw[h * 4 + g] = sc[g] * inv;
    }
    __syncthreads();

    for (int d = tid; d < D; d += 256) {
        float vv[4];
#pragma unroll
        for (int g = 0; g < 4; ++g) vv[g] = v[base + g * D + d];
#pragma unroll
        for (int h = 0; h < 4; ++h) {
            float o = attnw[h * 4 + 0] * vv[0] + attnw[h * 4 + 1] * vv[1] +
                      attnw[h * 4 + 2] * vv[2] + attnw[h * 4 + 3] * vv[3];
            aout[(size_t)((b * 4 + h) * S + s) * D + d] = f2bf(o);
        }
    }
}

// ---------------------------------------------------------------- launch
extern "C" void kernel_launch(void* const* d_in, const int* in_sizes, int n_in,
                              void* d_out, int out_size, void* d_ws, size_t ws_size,
                              hipStream_t stream) {
    (void)in_sizes; (void)n_in; (void)out_size; (void)ws_size;
    const int Bb = 4, S = 2048, Dm = 1024, HD = 4096, DOUT = 1024;
    const int M = Bb * S;  // 8192

    const float* x     = (const float*)d_in[0];
    const float* phase = (const float*)d_in[1];
    const float* Wq    = (const float*)d_in[2];
    const float* bq    = (const float*)d_in[3];
    const float* Wk    = (const float*)d_in[4];
    const float* bk    = (const float*)d_in[5];
    const float* Wv    = (const float*)d_in[6];
    const float* bv    = (const float*)d_in[7];
    const float* Wo    = (const float*)d_in[8];
    const float* bo    = (const float*)d_in[9];
    float* out = (float*)d_out;

    char* w = (char*)d_ws;
    unsigned short* x_bf = (unsigned short*)w; w += (size_t)M * Dm * 2;
    unsigned short* WqT  = (unsigned short*)w; w += (size_t)HD * Dm * 2;
    unsigned short* WkT  = (unsigned short*)w; w += (size_t)HD * Dm * 2;
    unsigned short* WvT  = (unsigned short*)w; w += (size_t)HD * Dm * 2;
    unsigned short* WoT  = (unsigned short*)w; w += (size_t)DOUT * HD * 2;
    float* qf = (float*)w; w += (size_t)M * HD * 4;
    float* kf = (float*)w; w += (size_t)M * HD * 4;
    float* vf = (float*)w; w += (size_t)M * HD * 4;
    unsigned short* aoutb = (unsigned short*)w; w += (size_t)M * HD * 2;

    convert_f32_bf16<<<(M * Dm + 255) / 256, 256, 0, stream>>>(x, x_bf, M * Dm);
    transpose_convert<<<dim3(HD / 32, Dm / 32), 256, 0, stream>>>(Wq, WqT, Dm, HD);
    transpose_convert<<<dim3(HD / 32, Dm / 32), 256, 0, stream>>>(Wk, WkT, Dm, HD);
    transpose_convert<<<dim3(HD / 32, Dm / 32), 256, 0, stream>>>(Wv, WvT, Dm, HD);
    transpose_convert<<<dim3(DOUT / 32, HD / 32), 256, 0, stream>>>(Wo, WoT, HD, DOUT);

    dim3 gqkv(HD / 64, M / 128);
    gemm_bf16_wmma<<<gqkv, 256, 0, stream>>>(x_bf, WqT, bq, qf, M, HD, Dm);
    gemm_bf16_wmma<<<gqkv, 256, 0, stream>>>(x_bf, WkT, bk, kf, M, HD, Dm);
    gemm_bf16_wmma<<<gqkv, 256, 0, stream>>>(x_bf, WvT, bv, vf, M, HD, Dm);

    attn_kernel<<<M, 256, 0, stream>>>(qf, kf, vf, phase, aoutb, S, Dm);

    gemm_bf16_wmma<<<dim3(DOUT / 64, M / 128), 256, 0, stream>>>(aoutb, WoT, bo, out, M, DOUT, HD);
}
